// MultiHeadAttention_86603720557256
// MI455X (gfx1250) — matrix operand order repaired and checked by reading, not compiled, not run on hardware
//
#include <hip/hip_runtime.h>

typedef __attribute__((ext_vector_type(16))) __bf16 v16bf;
typedef __attribute__((ext_vector_type(8)))  __bf16 v8bf;
typedef __attribute__((ext_vector_type(8)))  float  v8f;
typedef __attribute__((ext_vector_type(4)))  unsigned v4u;
typedef __attribute__((ext_vector_type(8)))  unsigned v8u;
typedef __attribute__((ext_vector_type(4)))  int      v4i;

#define DEV __device__ __forceinline__

#if defined(__has_builtin)
#if __has_builtin(__builtin_amdgcn_global_load_async_to_lds_b128)
#define HAVE_ASYNC_B128 1
#endif
#if __has_builtin(__builtin_amdgcn_s_wait_asynccnt)
#define HAVE_WAIT_ASYNC 1
#endif
#if __has_builtin(__builtin_amdgcn_s_wait_tensorcnt)
#define HAVE_WAIT_TENSOR 1
#endif
#endif

static constexpr int Bc = 4, Sc = 2048, Dc = 512, Hc = 8, Ec = 64;
static constexpr int MROWS = Bc * Sc;                 // 8192
static constexpr float SCALE = 0.04419417382415922f;  // 512^-0.5 (d_model, per reference)

DEV __bf16 f2bf(float f) {
  union { float f; unsigned u; } in; in.f = f;
  unsigned u = in.u;
  u += 0x7FFFu + ((u >> 16) & 1u);              // round-to-nearest-even
  union { unsigned short s; __bf16 b; } out; out.s = (unsigned short)(u >> 16);
  return out.b;
}

DEV v16bf cat16(v8bf a, v8bf b) {
  return __builtin_shufflevector(a, b, 0,1,2,3,4,5,6,7,8,9,10,11,12,13,14,15);
}

// ---- A-matrix fragment (16x32 bf16, ISA 7.12.2): lane l<16 holds row l,
// K = {k0..k0+7, k0+16..k0+23}; lanes 16-31 hold same rows, other K-halves.
DEV v16bf load_a(const __bf16* p, int ld, int row0, int k0) {
  int lane = threadIdx.x & 31;
  int r  = row0 + (lane & 15);
  int hv = (lane >> 4) & 1;
  const __bf16* q = p + (size_t)r * ld + k0 + hv * 8;
  v8bf lo = *(const v8bf*)q;
  v8bf hi = *(const v8bf*)(q + 16);
  return cat16(lo, hi);
}

// ---- B-matrix fragment (32x16 bf16) loaded from B^T stored row-major [N,K]:
// lane n (0-15) holds column col0+n with K = {k0..k0+7, k0+16..k0+23}; lanes 16-31 other K-halves.
DEV v16bf load_bT(const __bf16* bt, int ld, int col0, int k0) {
  int lane = threadIdx.x & 31;
  int c  = col0 + (lane & 15);
  int hv = (lane >> 4) & 1;
  const __bf16* q = bt + (size_t)c * ld + k0 + hv * 8;
  v8bf lo = *(const v8bf*)q;
  v8bf hi = *(const v8bf*)(q + 16);
  return cat16(lo, hi);
}

DEV v8f wmma_bf16(v16bf a, v16bf b, v8f c) {
  return __builtin_amdgcn_wmma_f32_16x16x32_bf16(false, a, false, b, (short)0, c,
                                                 false, false);
}

DEV float redmax16(float v) {
  v = fmaxf(v, __shfl_xor(v, 1, 32));
  v = fmaxf(v, __shfl_xor(v, 2, 32));
  v = fmaxf(v, __shfl_xor(v, 4, 32));
  v = fmaxf(v, __shfl_xor(v, 8, 32));
  return v;
}
DEV float redsum16(float v) {
  v += __shfl_xor(v, 1, 32);
  v += __shfl_xor(v, 2, 32);
  v += __shfl_xor(v, 4, 32);
  v += __shfl_xor(v, 8, 32);
  return v;
}

DEV void wait_async0() {
#if defined(HAVE_WAIT_ASYNC)
  __builtin_amdgcn_s_wait_asynccnt(0);
#else
  asm volatile("s_wait_asynccnt 0x0" ::: "memory");
#endif
}
DEV void wait_tensor0() {
#if defined(HAVE_WAIT_TENSOR)
  __builtin_amdgcn_s_wait_tensorcnt(0);
#else
  asm volatile("s_wait_tensorcnt 0x0" ::: "memory");
#endif
}

// ---------------- pack / convert kernels ----------------
__global__ void mha_convert_x(const float* __restrict__ x, __bf16* __restrict__ xbf) {
  int i = blockIdx.x * 256 + threadIdx.x;
  xbf[i] = f2bf(x[i]);
}

// wt[c*512 + k] = W_mat[h][k][e], c = mat*512 + h*64 + e  (B^T layout [N=1536, K=512])
__global__ void mha_pack_wqkv(const float* __restrict__ Wq, const float* __restrict__ Wk,
                              const float* __restrict__ Wv, __bf16* __restrict__ wt) {
  int i = blockIdx.x * 256 + threadIdx.x;          // 1536*512
  int c = i >> 9, kk = i & 511;
  int mat = c >> 9;
  int he  = c & 511;
  int h = he >> 6, e = he & 63;
  const float* W = (mat == 0) ? Wq : (mat == 1) ? Wk : Wv;
  wt[i] = f2bf(W[((size_t)h * Dc + kk) * Ec + e]);
}

// wot[n*512 + k] = Wo[k*512 + n]
__global__ void mha_pack_wo(const float* __restrict__ Wo, __bf16* __restrict__ wot) {
  int i = blockIdx.x * 256 + threadIdx.x;          // 512*512
  int n = i >> 9, kk = i & 511;
  wot[i] = f2bf(Wo[(size_t)kk * Dc + n]);
}

// ---------------- fused QKV projection GEMM ----------------
// [8192,512]bf16 x [512,1536] -> Q/K/V [B,H,S,E] bf16 (+bias)
// One wave computes a 16x64 tile: A fragment reused across 4 WMMAs per k-step.
__global__ __launch_bounds__(256) void mha_qkv_gemm(
    const __bf16* __restrict__ xbf, const __bf16* __restrict__ wt,
    const float* __restrict__ bq, const float* __restrict__ bk,
    const float* __restrict__ bv,
    __bf16* __restrict__ qb, __bf16* __restrict__ kbuf, __bf16* __restrict__ vbuf) {
  const int TN = 24;                                // 1536/64
  int wave = threadIdx.x >> 5;
  int lane = threadIdx.x & 31;
  int tile = blockIdx.x * 8 + wave;
  int tn = tile % TN, tm = tile / TN;
  int row0 = tm * 16, col0 = tn * 64;
  int n = lane & 15, hv = lane >> 4;

  v8f acc[4] = {};
#pragma unroll 2
  for (int k0 = 0; k0 < Dc; k0 += 32) {
    __builtin_prefetch(&xbf[(size_t)(row0 + n) * Dc + k0 + 64], 0, 0);
    v16bf a = load_a(xbf, Dc, row0, k0);
#pragma unroll
    for (int j = 0; j < 4; ++j) {
      v16bf b = load_bT(wt, Dc, col0 + j * 16, k0);
      acc[j] = wmma_bf16(a, b, acc[j]);
    }
  }

#pragma unroll
  for (int j = 0; j < 4; ++j) {
    int c   = col0 + j * 16 + n;                    // 0..1535
    int mat = c >> 9;
    int he  = c & 511;
    int h = he >> 6, e = he & 63;
    const float* bias = (mat == 0) ? bq : (mat == 1) ? bk : bv;
    float bi = bias[he];
    __bf16* outb = (mat == 0) ? qb : (mat == 1) ? kbuf : vbuf;
#pragma unroll
    for (int r = 0; r < 8; ++r) {
      int m = row0 + r + hv * 8;                    // global row = b*S + s
      int bI = m >> 11, s = m & (Sc - 1);
      outb[(((size_t)bI * Hc + h) * Sc + s) * Ec + e] = f2bf(acc[j][r] + bi);
    }
  }
}

// ---------------- FlashAttention (causal, online softmax) ----------------
// K block staged by the Tensor Data Mover; V block via async global->LDS copy,
// then transposed LDS->LDS for the B-fragment layout of P·V.
__global__ __launch_bounds__(256) void mha_flash_attn(
    const __bf16* __restrict__ qbuf, const __bf16* __restrict__ kbuf,
    const __bf16* __restrict__ vbuf, __bf16* __restrict__ attn_out) {
  __shared__ __attribute__((aligned(16))) __bf16 ldsK[32 * 64];    // K block [t][e]
  __shared__ __attribute__((aligned(16))) __bf16 ldsVraw[32 * 64]; // V block [t][e]
  __shared__ __attribute__((aligned(16))) __bf16 ldsVT[64 * 32];   // V^T [e][t]
  __shared__ __attribute__((aligned(16))) __bf16 ldsP[8][16 * 32]; // per-wave P staging

  int bh = blockIdx.y;
  int b = bh >> 3, h = bh & 7;
  int qblk = blockIdx.x;
  int wave = threadIdx.x >> 5;
  int lane = threadIdx.x & 31;
  int n = lane & 15, hv = lane >> 4;
  int qs = qblk * 128 + wave * 16;                  // this wave's 16 query rows

  size_t base = (size_t)bh * Sc * Ec;
  const __bf16* qp = qbuf + base;
  const __bf16* kp = kbuf + base;
  const __bf16* vp = vbuf + base;

  v16bf aq0 = load_a(qp, Ec, qs, 0);
  v16bf aq1 = load_a(qp, Ec, qs, 32);

  v8f o[4] = {};
  float m8[8], l8[8];
#pragma unroll
  for (int r = 0; r < 8; ++r) { m8[r] = -__builtin_inff(); l8[r] = 0.f; }

  unsigned ldsKoff = (unsigned)(unsigned long long)
      (__attribute__((address_space(3))) __bf16*)&ldsK[0];

  int nkb = qblk * 4 + 4;                           // causal: key blocks of 32
  for (int kbI = 0; kbI < nkb; ++kbI) {
    int kt = kbI * 32;
    __syncthreads();                                // fence prior reads of ldsK/ldsVT

    // ---- K block: one TDM descriptor per workgroup (issued by wave 0) ----
    if (wave == 0) {
      unsigned long long ga = (unsigned long long)&kp[(size_t)kt * Ec];
      v4u g0;
      g0[0] = 1u;                                   // count=1, user mode
      g0[1] = ldsKoff;                              // lds_addr (bytes)
      g0[2] = (unsigned)(ga & 0xffffffffu);         // global_addr[31:0]
      g0[3] = (unsigned)((ga >> 32) & 0x1ffffffu) | (2u << 30);  // addr[56:32], type=2
      v8u g1;
      g1[0] = 1u << 16;                             // data_size = 2 bytes
      g1[1] = (unsigned)Ec << 16;                   // tensor_dim0 = 64 (lo16 @ bits 63:48)
      g1[2] = (unsigned)Sc << 16;                   // dim0 hi=0 | tensor_dim1 = 2048
      g1[3] = (unsigned)Ec << 16;                   // dim1 hi=0 | tile_dim0 = 64
      g1[4] = 32u;                                  // tile_dim1 = 32, tile_dim2 = 0
      g1[5] = (unsigned)Ec;                         // tensor_dim0_stride = 64
      g1[6] = 0u;
      g1[7] = 0u;
      asm volatile("tensor_load_to_lds %0, %1" : : "s"(g0), "s"(g1) : "memory");
    }

    // ---- V block: async global->LDS (row-major), 16B per thread ----
    {
      int i = threadIdx.x * 8;                      // 256 threads * 8 elems = 2048
      int t = i >> 6, e0 = i & 63;
#if defined(HAVE_ASYNC_B128)
      __builtin_amdgcn_global_load_async_to_lds_b128(
          (v4i*)&vp[(size_t)(kt + t) * Ec + e0],
          (v4i*)&ldsVraw[i], 0, 0);
      wait_async0();
#else
      *(v8bf*)&ldsVraw[i] = *(const v8bf*)&vp[(size_t)(kt + t) * Ec + e0];
#endif
    }
    if (wave == 0) wait_tensor0();
    __syncthreads();                                // ldsK + ldsVraw visible

    {   // LDS->LDS transpose of V for the B-fragment layout
      int i = threadIdx.x * 8;
      int t = i >> 6, e0 = i & 63;
      v8bf vv = *(const v8bf*)&ldsVraw[i];
#pragma unroll
      for (int j = 0; j < 8; ++j) ldsVT[(e0 + j) * 32 + t] = vv[j];
    }
    __syncthreads();                                // ldsVT visible
    if (kt > qs + 15) continue;                     // fully masked for this wave (uniform)

    // scores: two 16x16 tiles over keys [kt..kt+31]
    v8f c0 = {}, c1 = {};
#pragma unroll
    for (int ec = 0; ec < 2; ++ec) {
      v16bf a   = ec ? aq1 : aq0;
      v16bf bk0 = load_bT(ldsK, Ec, 0,  ec * 32);
      v16bf bk1 = load_bT(ldsK, Ec, 16, ec * 32);
      c0 = wmma_bf16(a, bk0, c0);
      c1 = wmma_bf16(a, bk1, c1);
    }

    // mask + online softmax (row r lives in VGPR r across one 16-lane half)
#pragma unroll
    for (int r = 0; r < 8; ++r) {
      int srow = qs + r + hv * 8;
      float s0 = c0[r] * SCALE;
      float s1 = c1[r] * SCALE;
      if (kt + n      > srow) s0 = -__builtin_inff();
      if (kt + 16 + n > srow) s1 = -__builtin_inff();
      float tmax = redmax16(fmaxf(s0, s1));
      float mnew = fmaxf(m8[r], tmax);
      float corr = (m8[r] == mnew) ? 1.0f : __expf(m8[r] - mnew);
      float p0 = (s0 == -__builtin_inff()) ? 0.f : __expf(s0 - mnew);
      float p1 = (s1 == -__builtin_inff()) ? 0.f : __expf(s1 - mnew);
      float rs = redsum16(p0 + p1);
      l8[r] = l8[r] * corr + rs;
      m8[r] = mnew;
      c0[r] = p0; c1[r] = p1;
      o[0][r] *= corr; o[1][r] *= corr; o[2][r] *= corr; o[3][r] *= corr;
    }

    // restage P (C-layout f32 -> row-major bf16 16x32), reload as A fragment
    __bf16* ps = ldsP[wave];
#pragma unroll
    for (int r = 0; r < 8; ++r) {
      int row = r + hv * 8;
      ps[row * 32 + n]      = f2bf(c0[r]);
      ps[row * 32 + 16 + n] = f2bf(c1[r]);
    }
    asm volatile("s_wait_dscnt 0" ::: "memory");    // same-wave LDS RAW
    v16bf ap = load_a(ps, 32, 0, 0);

    // O += P (16x32) x V (32x64)
#pragma unroll
    for (int j = 0; j < 4; ++j) {
      v16bf bvf = load_bT(ldsVT, 32, j * 16, 0);
      o[j] = wmma_bf16(ap, bvf, o[j]);
    }
  }

  // epilogue: O /= l, write concat-head layout [b, s, h*64+e] as bf16
#pragma unroll
  for (int r = 0; r < 8; ++r) {
    float inv = (l8[r] > 0.f) ? 1.0f / l8[r] : 0.f;
    int srow = qs + r + hv * 8;
#pragma unroll
    for (int j = 0; j < 4; ++j) {
      size_t oi = ((size_t)(b * Sc + srow)) * Dc + h * Ec + j * 16 + n;
      attn_out[oi] = f2bf(o[j][r] * inv);
    }
  }
}

// ---------------- output projection GEMM (16x64 per wave) ----------------
__global__ __launch_bounds__(256) void mha_proj_gemm(
    const __bf16* __restrict__ attn, const __bf16* __restrict__ wot,
    const float* __restrict__ bo, float* __restrict__ out) {
  int wave = threadIdx.x >> 5;
  int lane = threadIdx.x & 31;
  int tile = blockIdx.x * 8 + wave;                 // 512*8 tiles
  int tn = tile & 7, tm = tile >> 3;
  int row0 = tm * 16, col0 = tn * 64;
  int n = lane & 15, hv = lane >> 4;

  v8f acc[4] = {};
#pragma unroll 2
  for (int k0 = 0; k0 < Dc; k0 += 32) {
    __builtin_prefetch(&attn[(size_t)(row0 + n) * Dc + k0 + 64], 0, 0);
    v16bf a = load_a(attn, Dc, row0, k0);
#pragma unroll
    for (int j = 0; j < 4; ++j) {
      v16bf b = load_bT(wot, Dc, col0 + j * 16, k0);
      acc[j] = wmma_bf16(a, b, acc[j]);
    }
  }
#pragma unroll
  for (int j = 0; j < 4; ++j) {
    int c = col0 + j * 16 + n;
    float bi = bo[c];
#pragma unroll
    for (int r = 0; r < 8; ++r) {
      int m = row0 + r + hv * 8;
      out[(size_t)m * Dc + c] = acc[j][r] + bi;
    }
  }
}

// ---------------- host launcher ----------------
extern "C" void kernel_launch(void* const* d_in, const int* in_sizes, int n_in,
                              void* d_out, int out_size, void* d_ws, size_t ws_size,
                              hipStream_t stream) {
  const float* x  = (const float*)d_in[0];
  const float* Wq = (const float*)d_in[1];
  const float* bq = (const float*)d_in[2];
  const float* Wk = (const float*)d_in[3];
  const float* bk = (const float*)d_in[4];
  const float* Wv = (const float*)d_in[5];
  const float* bv = (const float*)d_in[6];
  const float* Wo = (const float*)d_in[7];
  const float* bo = (const float*)d_in[8];
  float* out = (float*)d_out;

  char* ws = (char*)d_ws;
  size_t off = 0;
  auto alloc = [&](size_t bytes) {
    char* p = ws + off;
    off += (bytes + 255) & ~(size_t)255;
    return p;
  };
  __bf16* xbf   = (__bf16*)alloc((size_t)MROWS * Dc * 2);
  __bf16* qb    = (__bf16*)alloc((size_t)Bc * Hc * Sc * Ec * 2);
  __bf16* kbuf  = (__bf16*)alloc((size_t)Bc * Hc * Sc * Ec * 2);
  __bf16* vbuf  = (__bf16*)alloc((size_t)Bc * Hc * Sc * Ec * 2);
  __bf16* attnb = (__bf16*)alloc((size_t)MROWS * Dc * 2);
  __bf16* wt    = (__bf16*)alloc((size_t)3 * Dc * Dc * 2);
  __bf16* wot   = (__bf16*)alloc((size_t)Dc * Dc * 2);
  (void)ws_size; (void)in_sizes; (void)n_in; (void)out_size;

  mha_convert_x<<<(MROWS * Dc) / 256, 256, 0, stream>>>(x, xbf);
  mha_pack_wqkv<<<(3 * Dc * Dc) / 256, 256, 0, stream>>>(Wq, Wk, Wv, wt);
  mha_pack_wo<<<(Dc * Dc) / 256, 256, 0, stream>>>(Wo, wot);

  // QKV: 512 x 24 (16x64) tiles, 8 waves per block
  mha_qkv_gemm<<<(512 * 24) / 8, 256, 0, stream>>>(xbf, wt, bq, bk, bv, qb, kbuf, vbuf);

  dim3 ag(Sc / 128, Bc * Hc);                       // (16, 32)
  mha_flash_attn<<<ag, 256, 0, stream>>>(qb, kbuf, vbuf, attnb);

  // Projection: 512 x 8 (16x64) tiles, 8 waves per block
  mha_proj_gemm<<<(512 * 8) / 8, 256, 0, stream>>>(attnb, wot, bo, out);
}
